// NerfHash_51780125721347
// MI455X (gfx1250) — compile-verified, matrix operand order checked by reading, not run on hardware
//
#include <hip/hip_runtime.h>
#include <hip/hip_bf16.h>
#include <math.h>

typedef __attribute__((ext_vector_type(16))) _Float16 v16h;
typedef __attribute__((ext_vector_type(8)))  _Float16 v8h;
typedef __attribute__((ext_vector_type(8)))  float    v8f;

#define L_LEVELS 24
#define HASH_CAP (1 << 18)
#define ACT_STRIDE 104              // halfs per LDS row, multiple of 8 (b128 aligned)
#define SAMPLES_PER_BLOCK 128
#define THREADS_PER_BLOCK 256       // 8 wave32

// ---- WMMA-fragment-swizzled weight blob layout in d_ws (halfs) ----
#define OFF_W1 0        // 64x64  (K=51 padded)
#define OFF_W2 4096     // 64x64
#define OFF_W3 8192     // 64x64
#define OFF_W4 12288    // 64x80  (N=65 padded)
#define OFF_R1 17408    // 96x64  (K=80 padded)
#define OFF_R2 23552    // 64x64
#define OFF_R3 27648    // 64x16  (N=3 padded)
#define WFRAG_TOTAL 28672

__device__ __forceinline__ void wave_lds_fence() {
  // intra-wave LDS RAW/WAR ordering (only this wave touches its rows)
  asm volatile("s_wait_dscnt 0x0" ::: "memory");
}

__device__ __forceinline__ float gelu_f(float x) {
  return 0.5f * x * (1.0f + erff(x * 0.70710678118654752f));
}
__device__ __forceinline__ float softplus_f(float x) {
  return fmaxf(x, 0.0f) + log1pf(expf(-fabsf(x)));
}
__device__ __forceinline__ float sigmoid_f(float x) {
  return 1.0f / (1.0f + expf(-x));
}

// ---------------- weight pre-swizzle: f32 row-major -> f16 B-fragment blob ----------------
__global__ void prep_weights(const float* __restrict__ W1, const float* __restrict__ W2,
                             const float* __restrict__ W3, const float* __restrict__ W4,
                             const float* __restrict__ R1, const float* __restrict__ R2,
                             const float* __restrict__ R3, _Float16* __restrict__ wf) {
  int i = blockIdx.x * blockDim.x + threadIdx.x;
  if (i >= WFRAG_TOTAL) return;
  const float* src; int K, Nout, Nt, base;
  if      (i < OFF_W2) { src = W1; K = 51; Nout = 64; Nt = 4; base = OFF_W1; }
  else if (i < OFF_W3) { src = W2; K = 64; Nout = 64; Nt = 4; base = OFF_W2; }
  else if (i < OFF_W4) { src = W3; K = 64; Nout = 64; Nt = 4; base = OFF_W3; }
  else if (i < OFF_R1) { src = W4; K = 64; Nout = 65; Nt = 5; base = OFF_W4; }
  else if (i < OFF_R2) { src = R1; K = 80; Nout = 64; Nt = 4; base = OFF_R1; }
  else if (i < OFF_R3) { src = R2; K = 64; Nout = 64; Nt = 4; base = OFF_R2; }
  else                 { src = R3; K = 64; Nout =  3; Nt = 1; base = OFF_R3; }
  int idx = i - base;
  int c  = idx >> 9;          // fragment (kc,nt) chunk: 512 halfs each
  int l  = (idx >> 4) & 31;   // lane
  int j  = idx & 15;          // half within lane
  int kc = c / Nt, nt = c % Nt;
  int h  = l >> 4;
  int n  = nt * 16 + (l & 15);
  int k  = kc * 32 + ((j >> 3) << 4) + h * 8 + (j & 7);   // ISA 16-bit B layout
  float v = (k < K && n < Nout) ? src[k * Nout + n] : 0.0f;
  wf[i] = (_Float16)v;
}

// ---------------- fragment loaders ----------------
__device__ __forceinline__ v16h load_a_frag(const _Float16* __restrict__ actRow, int kc, int h) {
  // lane owns row m = lane&15; halves j=0..7 -> K = kc*32 + h*8 + j ; j=8..15 -> +16
  const _Float16* p = actRow + kc * 32 + h * 8;
  v8h lo = *(const v8h*)(p);
  v8h hi = *(const v8h*)(p + 16);
  return __builtin_shufflevector(lo, hi, 0,1,2,3,4,5,6,7,8,9,10,11,12,13,14,15);
}

__device__ __forceinline__ v16h load_b_frag(const _Float16* __restrict__ wf, int base,
                                            int kc, int nt, int Nt, int lane) {
  const _Float16* p = wf + base + (size_t)(kc * Nt + nt) * 512 + lane * 16;
  v8h lo = *(const v8h*)(p);
  v8h hi = *(const v8h*)(p + 8);
  return __builtin_shufflevector(lo, hi, 0,1,2,3,4,5,6,7,8,9,10,11,12,13,14,15);
}

template <int KC, int NT>
__device__ __forceinline__ void layer_mm(const _Float16* __restrict__ wf, int wbase,
                                         const _Float16* __restrict__ actRow, int lane,
                                         v8f (&acc)[NT]) {
  const v8f zero = {0.f,0.f,0.f,0.f,0.f,0.f,0.f,0.f};
#pragma unroll
  for (int nt = 0; nt < NT; ++nt) acc[nt] = zero;
  int h = lane >> 4;
#pragma unroll
  for (int kc = 0; kc < KC; ++kc) {
    v16h a = load_a_frag(actRow, kc, h);
#pragma unroll
    for (int nt = 0; nt < NT; ++nt) {
      v16h b = load_b_frag(wf, wbase, kc, nt, NT, lane);
      acc[nt] = __builtin_amdgcn_wmma_f32_16x16x32_f16(
          false, a, false, b, (short)0, acc[nt], false, false);
    }
  }
}

// bias + gelu + store back to LDS (cols 0..NT*16-1)
template <int NT>
__device__ __forceinline__ void store_gelu(v8f (&acc)[NT], const float* __restrict__ bias,
                                           _Float16* __restrict__ actWave, int lane) {
  int h = lane >> 4, nl = lane & 15;
#pragma unroll
  for (int nt = 0; nt < NT; ++nt) {
    int nn = nt * 16 + nl;
    float b = bias[nn];
#pragma unroll
    for (int r = 0; r < 8; ++r) {
      int m = r + h * 8;
      actWave[m * ACT_STRIDE + nn] = (_Float16)gelu_f(acc[nt][r] + b);
    }
  }
}

// ---------------- fused NeRF kernel ----------------
__global__ __launch_bounds__(THREADS_PER_BLOCK, 2)
void nerf_fused(const float* __restrict__ pos, const float* __restrict__ dirs,
                const int* __restrict__ iter_nr,
                const float* __restrict__ tables, const int* __restrict__ shifts,
                const _Float16* __restrict__ wf,
                const float* __restrict__ b1, const float* __restrict__ b2,
                const float* __restrict__ b3, const float* __restrict__ b4,
                const float* __restrict__ rb1, const float* __restrict__ rb2,
                const float* __restrict__ rb3,
                float* __restrict__ out_rgb, float* __restrict__ out_density, int n) {
  __shared__ __align__(16) _Float16 actS[SAMPLES_PER_BLOCK * ACT_STRIDE];

  const int tid  = threadIdx.x;
  const int wave = tid >> 5;
  const int lane = tid & 31;
  const int s    = lane & 15;
  const int h    = lane >> 4;

  _Float16* actWave = actS + (wave * 16) * ACT_STRIDE;
  _Float16* actRow  = actWave + s * ACT_STRIDE;

  const int gs0 = blockIdx.x * SAMPLES_PER_BLOCK + wave * 16;  // wave tile base sample
  int gs = gs0 + s;
  if (gs >= n) gs = n - 1;  // clamp: keep EXEC all-ones for WMMA

  // coarse-to-fine window parameter
  float t = 0.3f + 0.7f * fminf(fmaxf((float)iter_nr[0] * (1.0f / 10000.0f), 0.0f), 1.0f);

  float px = pos[gs * 3 + 0];
  float py = pos[gs * 3 + 1];
  float pz = pos[gs * 3 + 2];

  // ---- hash-grid encode: 2 lanes per sample, 12 levels each ----
  for (int li = 0; li < 12; ++li) {
    int lev = h * 12 + li;
    // geomspace(1, 1e-4, 24): scale = exp(-ln(1e4)/23 * lev)  -> inv = exp(+...)
    float inv = expf(0.40045827704244276f * (float)lev);
    float cx = px * inv, cy = py * inv, cz = pz * inv;
    float flx = floorf(cx), fly = floorf(cy), flz = floorf(cz);
    float fx = cx - flx, fy = cy - fly, fz = cz - flz;
    int bx = (int)flx + shifts[lev * 3 + 0];
    int by = (int)fly + shifts[lev * 3 + 1];
    int bz = (int)flz + shifts[lev * 3 + 2];
    const float* tb = tables + (size_t)lev * HASH_CAP * 2;
    float f0 = 0.f, f1 = 0.f;
#pragma unroll
    for (int corner = 0; corner < 8; ++corner) {
      int ox = corner & 1, oy = (corner >> 1) & 1, oz = (corner >> 2) & 1;
      unsigned hx = (unsigned)(bx + ox);
      unsigned hy = (unsigned)(by + oy) * 2654435761u;
      unsigned hz = (unsigned)(bz + oz) * 805459861u;
      unsigned idx = (hx ^ hy ^ hz) & (unsigned)(HASH_CAP - 1);
      float2 f = *(const float2*)(tb + (size_t)idx * 2);
      float w = (ox ? fx : 1.f - fx) * (oy ? fy : 1.f - fy) * (oz ? fz : 1.f - fz);
      f0 += w * f.x;
      f1 += w * f.y;
    }
    float wl = 0.5f * (1.0f - cosf(3.14159265358979f *
                 fminf(fmaxf(t * (float)L_LEVELS - (float)lev, 0.0f), 1.0f)));
    actRow[lev * 2 + 0] = (_Float16)(f0 * wl);
    actRow[lev * 2 + 1] = (_Float16)(f1 * wl);
  }
  if (h == 0) {  // cols 48..50 = pos, 51..63 = 0 (zero-pad K for layer 1)
    v8h a = {};
    a[0] = (_Float16)px; a[1] = (_Float16)py; a[2] = (_Float16)pz;
    *(v8h*)(actRow + 48) = a;
    v8h z = {};
    *(v8h*)(actRow + 56) = z;
  }
  wave_lds_fence();

  // ---- density MLP: 3x (64->64) gelu ----
  v8f acc4[4];
  layer_mm<2, 4>(wf, OFF_W1, actRow, lane, acc4);
  wave_lds_fence();
  store_gelu<4>(acc4, b1, actWave, lane);
  wave_lds_fence();
  layer_mm<2, 4>(wf, OFF_W2, actRow, lane, acc4);
  wave_lds_fence();
  store_gelu<4>(acc4, b2, actWave, lane);
  wave_lds_fence();
  layer_mm<2, 4>(wf, OFF_W3, actRow, lane, acc4);
  wave_lds_fence();
  store_gelu<4>(acc4, b3, actWave, lane);
  wave_lds_fence();

  // ---- fd = h @ W4 + b4 (65 outputs, padded to 80) ----
  v8f acc5[5];
  layer_mm<2, 5>(wf, OFF_W4, actRow, lane, acc5);
  wave_lds_fence();
  {
    int nl = lane & 15;
#pragma unroll
    for (int nt = 0; nt < 5; ++nt) {
      int col = nt * 16 + nl;
      float b = (col < 65) ? b4[col] : 0.0f;
#pragma unroll
      for (int r = 0; r < 8; ++r) {
        int m = r + h * 8;
        float v = acc5[nt][r] + b;
        if (col == 0) {
          int gd = gs0 + m; if (gd >= n) gd = n - 1;
          out_density[gd] = softplus_f(v);
        } else if (col < 65) {
          actWave[m * ACT_STRIDE + (col - 1)] = (_Float16)gelu_f(v);  // hr cols 0..63
        }
      }
    }
  }

  // ---- SH degree-4 basis into cols 64..79, zero pad 80..95 ----
  {
    float dx = dirs[gs * 3 + 0], dy = dirs[gs * 3 + 1], dz = dirs[gs * 3 + 2];
    float xx = dx * dx, yy = dy * dy, zz = dz * dz;
    float sh[16];
    sh[0]  = 0.28209479177387814f;
    sh[1]  = -0.48860251190291987f * dy;
    sh[2]  =  0.48860251190291987f * dz;
    sh[3]  = -0.48860251190291987f * dx;
    sh[4]  =  1.0925484305920792f * dx * dy;
    sh[5]  = -1.0925484305920792f * dy * dz;
    sh[6]  =  0.94617469575756f * zz - 0.31539156525252f;
    sh[7]  = -1.0925484305920792f * dx * dz;
    sh[8]  =  0.5462742152960396f * (xx - yy);
    sh[9]  = -0.5900435899266435f * dy * (3.0f * xx - yy);
    sh[10] =  2.890611442640554f * dx * dy * dz;
    sh[11] = -0.4570457994644657f * dy * (4.0f * zz - xx - yy);
    sh[12] =  0.37317633259011546f * dz * (2.0f * zz - 3.0f * xx - 3.0f * yy);
    sh[13] = -0.4570457994644657f * dx * (4.0f * zz - xx - yy);
    sh[14] =  1.445305721320277f * dz * (xx - yy);
    sh[15] = -0.5900435899266435f * dx * (xx - 3.0f * yy);
#pragma unroll
    for (int j = 0; j < 8; ++j) actRow[64 + h * 8 + j] = (_Float16)sh[h * 8 + j];
#pragma unroll
    for (int j = 0; j < 8; ++j) actRow[80 + h * 8 + j] = (_Float16)0.0f;
  }
  wave_lds_fence();

  // ---- RGB MLP: (80->64) gelu, (64->64) gelu, (64->3) sigmoid ----
  layer_mm<3, 4>(wf, OFF_R1, actRow, lane, acc4);
  wave_lds_fence();
  store_gelu<4>(acc4, rb1, actWave, lane);
  wave_lds_fence();
  layer_mm<2, 4>(wf, OFF_R2, actRow, lane, acc4);
  wave_lds_fence();
  store_gelu<4>(acc4, rb2, actWave, lane);
  wave_lds_fence();

  v8f acc1[1];
  layer_mm<2, 1>(wf, OFF_R3, actRow, lane, acc1);
  {
    int nl = lane & 15;
    if (nl < 3) {
      float b = rb3[nl];
#pragma unroll
      for (int r = 0; r < 8; ++r) {
        int m = r + h * 8;
        int gd = gs0 + m; if (gd >= n) gd = n - 1;
        out_rgb[(size_t)gd * 3 + nl] = sigmoid_f(acc1[0][r] + b);
      }
    }
  }
}

extern "C" void kernel_launch(void* const* d_in, const int* in_sizes, int n_in,
                              void* d_out, int out_size, void* d_ws, size_t ws_size,
                              hipStream_t stream) {
  const float* pos    = (const float*)d_in[0];
  const float* dirs   = (const float*)d_in[1];
  const int*   iter   = (const int*)d_in[2];
  const float* tables = (const float*)d_in[3];
  const int*   shifts = (const int*)d_in[4];
  const float* W1 = (const float*)d_in[5],  *b1  = (const float*)d_in[6];
  const float* W2 = (const float*)d_in[7],  *b2  = (const float*)d_in[8];
  const float* W3 = (const float*)d_in[9],  *b3  = (const float*)d_in[10];
  const float* W4 = (const float*)d_in[11], *b4  = (const float*)d_in[12];
  const float* R1 = (const float*)d_in[13], *rb1 = (const float*)d_in[14];
  const float* R2 = (const float*)d_in[15], *rb2 = (const float*)d_in[16];
  const float* R3 = (const float*)d_in[17], *rb3 = (const float*)d_in[18];

  const int n = in_sizes[0] / 3;
  _Float16* wf = (_Float16*)d_ws;
  float* out_rgb = (float*)d_out;
  float* out_density = out_rgb + (size_t)n * 3;

  prep_weights<<<(WFRAG_TOTAL + 255) / 256, 256, 0, stream>>>(W1, W2, W3, W4, R1, R2, R3, wf);

  int blocks = (n + SAMPLES_PER_BLOCK - 1) / SAMPLES_PER_BLOCK;
  nerf_fused<<<blocks, THREADS_PER_BLOCK, 0, stream>>>(
      pos, dirs, iter, tables, shifts, wf,
      b1, b2, b3, b4, rb1, rb2, rb3, out_rgb, out_density, n);
}
